// Multiscale_MessagePassing_Layer_27419071217667
// MI455X (gfx1250) — matrix operand order repaired and checked by reading, not compiled, not run on hardware
//
#include <hip/hip_runtime.h>
#include <hip/hip_bf16.h>

typedef __attribute__((ext_vector_type(16))) _Float16 v16h;
typedef __attribute__((ext_vector_type(8)))  _Float16 v8h;
typedef __attribute__((ext_vector_type(4)))  _Float16 v4h;
typedef __attribute__((ext_vector_type(8)))  float    v8f;

#define Hdim   128
#define LN_EPS 1e-5f

__device__ __forceinline__ float elu_f(float x) {
    return x > 0.f ? x : (expf(x) - 1.f);
}

// Load a 16x32 f16 A-fragment from an LDS tile (row-major, stride in f16 elems).
// Contiguous K layout (matches packed B): lanes 0-15 -> K {0..15}; lanes 16-31 -> K {16..31}.
__device__ __forceinline__ v16h load_a_frag(const _Float16* tile, int stride, int k0, int lane) {
    const int row  = lane & 15;
    const int koff = (lane >> 4) << 4;
    const _Float16* p = tile + row * stride + k0 + koff;
    v8h lo = *(const v8h*)p;
    v8h hi = *(const v8h*)(p + 8);
    return __builtin_shufflevector(lo, hi, 0,1,2,3,4,5,6,7,8,9,10,11,12,13,14,15);
}

// Load a 32x16 f16 B-fragment from pre-packed weights: per lane, 16 contiguous f16
// (col = col0 + lane&15, K half selected by lane>>4) -> one 32-byte load.
__device__ __forceinline__ v16h load_b_frag(const _Float16* wp, int ks, int col0, int lane) {
    const int col = col0 + (lane & 15);
    const int khi = lane >> 4;
    return *(const v16h*)(wp + ((size_t)((ks * Hdim + col) * 2 + khi) << 4));
}

__global__ void zero_f32_kernel(float* __restrict__ p, long long n) {
    long long i = (long long)blockIdx.x * blockDim.x + threadIdx.x;
    const long long stride = (long long)gridDim.x * blockDim.x;
    for (; i < n; i += stride) p[i] = 0.f;
}

__global__ void deg_kernel(const int* __restrict__ tgt, float* __restrict__ deg, int E) {
    int i = blockIdx.x * blockDim.x + threadIdx.x;
    if (i < E) atomicAdd(&deg[tgt[i]], 1.0f);
}

__global__ void invdeg_kernel(float* __restrict__ deg, int n) {
    int i = blockIdx.x * blockDim.x + threadIdx.x;
    if (i < n) {
        float d = deg[i];
        deg[i] = d > 0.f ? 1.f / d : 0.f;
    }
}

// Repack f32 weight [iters][K][128] into per-lane-contiguous f16 WMMA B-fragment order.
__global__ void pack_w_kernel(const float* __restrict__ w, _Float16* __restrict__ dst,
                              int K, int total) {
    int idx = blockIdx.x * blockDim.x + threadIdx.x;
    if (idx >= total) return;
    const int per = K * Hdim;
    const int it  = idx / per;
    const int r   = idx - it * per;
    const int k   = r / Hdim;
    const int col = r - k * Hdim;
    const int kb  = k >> 5;
    const int khi = (k >> 4) & 1;
    const int kk  = k & 15;
    dst[(size_t)it * per + (((kb * Hdim + col) * 2 + khi) * 16 + kk)] = (_Float16)w[idx];
}

// Edge update, 32 edges per block:
//   e = ELU([x[src]|x[tgt]|ea] @ W0 + b0) @ W1 + b1; ea = LN(ea + e); agg[tgt] += ea
__global__ __launch_bounds__(256)
void edge_kernel(const float* __restrict__ x,
                 const int* __restrict__ src, const int* __restrict__ tgt,
                 float* __restrict__ ea, float* __restrict__ agg,
                 const _Float16* __restrict__ w0p, const float* __restrict__ b0,
                 const _Float16* __restrict__ w1p, const float* __restrict__ b1,
                 const float* __restrict__ lng, const float* __restrict__ lnb)
{
    __shared__ __align__(16) _Float16 At[32][392];   // 3H + 8 pad
    __shared__ __align__(16) _Float16 Mid[32][136];  // H + 8 pad
    __shared__ __align__(16) float    Out[32][132];
    __shared__ int sS[32], sT[32];

    const int tid = threadIdx.x;
    const int e0  = blockIdx.x * 32;

    if (tid < 32)       sS[tid]      = src[e0 + tid];
    else if (tid < 64)  sT[tid - 32] = tgt[e0 + tid - 32];
    __syncthreads();

    // Stage A = [x[src] | x[tgt] | ea] as f16, 32 x 384 (3072 float4 groups)
    #pragma unroll
    for (int it = 0; it < 12; ++it) {
        int flat = tid + it * 256;
        int r    = flat / 96;
        int c4   = (flat % 96) * 4;
        float4 v;
        if (c4 < 128)      v = *(const float4*)(x  + (size_t)sS[r] * Hdim + c4);
        else if (c4 < 256) v = *(const float4*)(x  + (size_t)sT[r] * Hdim + (c4 - 128));
        else               v = *(const float4*)(ea + (size_t)(e0 + r) * Hdim + (c4 - 256));
        v4h h;
        h[0] = (_Float16)v.x; h[1] = (_Float16)v.y;
        h[2] = (_Float16)v.z; h[3] = (_Float16)v.w;
        *(v4h*)&At[r][c4] = h;
    }
    __syncthreads();

    const int lane  = tid & 31;
    const int wave  = tid >> 5;
    const int col0  = wave << 4;
    const int colL  = col0 + (lane & 15);
    const int rbase = (lane >> 4) << 3;

    // GEMM1: [32,384] @ [384,128]; each wave: 2 row-tiles x 1 col-tile, B reused.
    v8f cA = {}, cB = {};
    #pragma unroll
    for (int ks = 0; ks < 12; ++ks) {
        v16h b  = load_b_frag(w0p, ks, col0, lane);
        v16h a0 = load_a_frag(&At[0][0],  392, ks * 32, lane);
        v16h a1 = load_a_frag(&At[16][0], 392, ks * 32, lane);
        cA = __builtin_amdgcn_wmma_f32_16x16x32_f16(false, a0, false, b, (short)0, cA, false, false);
        cB = __builtin_amdgcn_wmma_f32_16x16x32_f16(false, a1, false, b, (short)0, cB, false, false);
    }
    {
        float bz = b0[colL];
        #pragma unroll
        for (int j = 0; j < 8; ++j) {
            Mid[rbase + j][colL]      = (_Float16)elu_f(cA[j] + bz);
            Mid[16 + rbase + j][colL] = (_Float16)elu_f(cB[j] + bz);
        }
    }
    __syncthreads();

    // GEMM2: [32,128] @ [128,128], 4 K-steps
    v8f dA = {}, dB = {};
    #pragma unroll
    for (int ks = 0; ks < 4; ++ks) {
        v16h b  = load_b_frag(w1p, ks, col0, lane);
        v16h a0 = load_a_frag(&Mid[0][0],  136, ks * 32, lane);
        v16h a1 = load_a_frag(&Mid[16][0], 136, ks * 32, lane);
        dA = __builtin_amdgcn_wmma_f32_16x16x32_f16(false, a0, false, b, (short)0, dA, false, false);
        dB = __builtin_amdgcn_wmma_f32_16x16x32_f16(false, a1, false, b, (short)0, dB, false, false);
    }
    {
        float bz = b1[colL];
        #pragma unroll
        for (int j = 0; j < 8; ++j) {
            Out[rbase + j][colL]      = dA[j] + bz;
            Out[16 + rbase + j][colL] = dB[j] + bz;
        }
    }
    __syncthreads();

    // Residual + LayerNorm (8 threads per row, shuffle width 8) + scatter-add
    {
        const int row   = tid >> 3;          // 0..31
        const int cbase = (tid & 7) * 16;    // 16 cols per thread
        const float* earow = ea + (size_t)(e0 + row) * Hdim;
        float v[16];
        float s = 0.f, ss = 0.f;
        #pragma unroll
        for (int q = 0; q < 16; ++q) {
            float t = earow[cbase + q] + Out[row][cbase + q];
            v[q] = t; s += t; ss += t * t;
        }
        #pragma unroll
        for (int m = 1; m < 8; m <<= 1) {
            s  += __shfl_xor(s,  m, 8);
            ss += __shfl_xor(ss, m, 8);
        }
        const float mu   = s * (1.f / Hdim);
        const float var  = ss * (1.f / Hdim) - mu * mu;
        const float rstd = rsqrtf(var + LN_EPS);
        float* eout = ea  + (size_t)(e0 + row) * Hdim;
        float* arow = agg + (size_t)sT[row] * Hdim;
        #pragma unroll
        for (int q = 0; q < 16; ++q) {
            const int cc = cbase + q;
            float y = (v[q] - mu) * rstd * lng[cc] + lnb[cc];
            eout[cc] = y;
            atomicAdd(arow + cc, y);
        }
    }
}

// Node update, 32 nodes per block:
//   h = ELU([x | agg*invdeg] @ W0 + b0) @ W1 + b1; xout = LN(x + h)
__global__ __launch_bounds__(256)
void node_kernel(const float* __restrict__ x,
                 const float* __restrict__ agg, const float* __restrict__ invdeg,
                 const _Float16* __restrict__ w0p, const float* __restrict__ b0,
                 const _Float16* __restrict__ w1p, const float* __restrict__ b1,
                 const float* __restrict__ lng, const float* __restrict__ lnb,
                 float* __restrict__ xout)
{
    __shared__ __align__(16) _Float16 At[32][264];   // 2H + 8 pad
    __shared__ __align__(16) _Float16 Mid[32][136];
    __shared__ __align__(16) float    Out[32][132];

    const int tid = threadIdx.x;
    const int n0  = blockIdx.x * 32;

    // Stage A = [x | agg * invdeg], 32 x 256 (2048 float4 groups)
    #pragma unroll
    for (int it = 0; it < 8; ++it) {
        int flat = tid + it * 256;
        int r    = flat >> 6;
        int c4   = (flat & 63) * 4;
        float4 v;
        if (c4 < 128) {
            v = *(const float4*)(x + (size_t)(n0 + r) * Hdim + c4);
        } else {
            float idg = invdeg[n0 + r];
            float4 a  = *(const float4*)(agg + (size_t)(n0 + r) * Hdim + (c4 - 128));
            v.x = a.x * idg; v.y = a.y * idg; v.z = a.z * idg; v.w = a.w * idg;
        }
        v4h h;
        h[0] = (_Float16)v.x; h[1] = (_Float16)v.y;
        h[2] = (_Float16)v.z; h[3] = (_Float16)v.w;
        *(v4h*)&At[r][c4] = h;
    }
    __syncthreads();

    const int lane  = tid & 31;
    const int wave  = tid >> 5;
    const int col0  = wave << 4;
    const int colL  = col0 + (lane & 15);
    const int rbase = (lane >> 4) << 3;

    v8f cA = {}, cB = {};
    #pragma unroll
    for (int ks = 0; ks < 8; ++ks) {
        v16h b  = load_b_frag(w0p, ks, col0, lane);
        v16h a0 = load_a_frag(&At[0][0],  264, ks * 32, lane);
        v16h a1 = load_a_frag(&At[16][0], 264, ks * 32, lane);
        cA = __builtin_amdgcn_wmma_f32_16x16x32_f16(false, a0, false, b, (short)0, cA, false, false);
        cB = __builtin_amdgcn_wmma_f32_16x16x32_f16(false, a1, false, b, (short)0, cB, false, false);
    }
    {
        float bz = b0[colL];
        #pragma unroll
        for (int j = 0; j < 8; ++j) {
            Mid[rbase + j][colL]      = (_Float16)elu_f(cA[j] + bz);
            Mid[16 + rbase + j][colL] = (_Float16)elu_f(cB[j] + bz);
        }
    }
    __syncthreads();

    v8f dA = {}, dB = {};
    #pragma unroll
    for (int ks = 0; ks < 4; ++ks) {
        v16h b  = load_b_frag(w1p, ks, col0, lane);
        v16h a0 = load_a_frag(&Mid[0][0],  136, ks * 32, lane);
        v16h a1 = load_a_frag(&Mid[16][0], 136, ks * 32, lane);
        dA = __builtin_amdgcn_wmma_f32_16x16x32_f16(false, a0, false, b, (short)0, dA, false, false);
        dB = __builtin_amdgcn_wmma_f32_16x16x32_f16(false, a1, false, b, (short)0, dB, false, false);
    }
    {
        float bz = b1[colL];
        #pragma unroll
        for (int j = 0; j < 8; ++j) {
            Out[rbase + j][colL]      = dA[j] + bz;
            Out[16 + rbase + j][colL] = dB[j] + bz;
        }
    }
    __syncthreads();

    {
        const int row   = tid >> 3;
        const int cbase = (tid & 7) * 16;
        const float* xrow = x + (size_t)(n0 + row) * Hdim;
        float v[16];
        float s = 0.f, ss = 0.f;
        #pragma unroll
        for (int q = 0; q < 16; ++q) {
            float t = xrow[cbase + q] + Out[row][cbase + q];
            v[q] = t; s += t; ss += t * t;
        }
        #pragma unroll
        for (int m = 1; m < 8; m <<= 1) {
            s  += __shfl_xor(s,  m, 8);
            ss += __shfl_xor(ss, m, 8);
        }
        const float mu   = s * (1.f / Hdim);
        const float var  = ss * (1.f / Hdim) - mu * mu;
        const float rstd = rsqrtf(var + LN_EPS);
        float* orow = xout + (size_t)(n0 + row) * Hdim;
        #pragma unroll
        for (int q = 0; q < 16; ++q) {
            const int cc = cbase + q;
            orow[cc] = (v[q] - mu) * rstd * lng[cc] + lnb[cc];
        }
    }
}

extern "C" void kernel_launch(void* const* d_in, const int* in_sizes, int n_in,
                              void* d_out, int out_size, void* d_ws, size_t ws_size,
                              hipStream_t stream)
{
    const float* x_in  = (const float*)d_in[0];
    const int*   eidx  = (const int*)d_in[1];
    const float* ea_in = (const float*)d_in[2];
    const float* ew0 = (const float*)d_in[5];
    const float* eb0 = (const float*)d_in[6];
    const float* ew1 = (const float*)d_in[7];
    const float* eb1 = (const float*)d_in[8];
    const float* elg = (const float*)d_in[9];
    const float* elb = (const float*)d_in[10];
    const float* nw0 = (const float*)d_in[11];
    const float* nb0 = (const float*)d_in[12];
    const float* nw1 = (const float*)d_in[13];
    const float* nb1 = (const float*)d_in[14];
    const float* nlg = (const float*)d_in[15];
    const float* nlb = (const float*)d_in[16];
    (void)n_in; (void)out_size;

    const int N   = in_sizes[0] / Hdim;   // 100000
    const int E   = in_sizes[2] / Hdim;   // 600000
    const int NMP = in_sizes[6] / Hdim;   // 4

    const int* src = eidx;
    const int* tgt = eidx + E;

    // Workspace carve-up (256B aligned)
    char*  base = (char*)d_ws;
    size_t off  = 0;
    auto take = [&](size_t bytes) -> char* {
        char* p = base + off;
        off = (off + bytes + 255) & ~(size_t)255;
        return p;
    };
    float*    ea   = (float*)take((size_t)E * Hdim * 4);
    float*    xws  = (float*)take((size_t)N * Hdim * 4);
    float*    agg  = (float*)take((size_t)N * Hdim * 4);
    float*    idg  = (float*)take((size_t)N * 4);
    _Float16* we0p = (_Float16*)take((size_t)NMP * 3 * Hdim * Hdim * 2);
    _Float16* we1p = (_Float16*)take((size_t)NMP * Hdim * Hdim * 2);
    _Float16* wn0p = (_Float16*)take((size_t)NMP * 2 * Hdim * Hdim * 2);
    _Float16* wn1p = (_Float16*)take((size_t)NMP * Hdim * Hdim * 2);
    (void)ws_size;

    // edge_attr working copy (inputs must stay untouched)
    hipMemcpyAsync(ea, ea_in, (size_t)E * Hdim * 4, hipMemcpyDeviceToDevice, stream);

    // degree -> inverse degree
    zero_f32_kernel<<<512, 256, 0, stream>>>(idg, (long long)N);
    deg_kernel<<<(E + 255) / 256, 256, 0, stream>>>(tgt, idg, E);
    invdeg_kernel<<<(N + 255) / 256, 256, 0, stream>>>(idg, N);

    // pack weights into WMMA B-fragment layout (f16)
    pack_w_kernel<<<(NMP * 3 * Hdim * Hdim + 255) / 256, 256, 0, stream>>>(ew0, we0p, 3 * Hdim, NMP * 3 * Hdim * Hdim);
    pack_w_kernel<<<(NMP * Hdim * Hdim + 255) / 256, 256, 0, stream>>>(ew1, we1p, Hdim, NMP * Hdim * Hdim);
    pack_w_kernel<<<(NMP * 2 * Hdim * Hdim + 255) / 256, 256, 0, stream>>>(nw0, wn0p, 2 * Hdim, NMP * 2 * Hdim * Hdim);
    pack_w_kernel<<<(NMP * Hdim * Hdim + 255) / 256, 256, 0, stream>>>(nw1, wn1p, Hdim, NMP * Hdim * Hdim);

    const float* xcur = x_in;
    for (int i = 0; i < NMP; ++i) {
        float* xnext = (i & 1) ? (float*)d_out : xws;   // i=3 writes d_out
        zero_f32_kernel<<<2048, 256, 0, stream>>>(agg, (long long)N * Hdim);
        edge_kernel<<<E / 32, 256, 0, stream>>>(
            xcur, src, tgt, ea, agg,
            we0p + (size_t)i * 3 * Hdim * Hdim, eb0 + i * Hdim,
            we1p + (size_t)i * Hdim * Hdim,     eb1 + i * Hdim,
            elg + i * Hdim, elb + i * Hdim);
        node_kernel<<<N / 32, 256, 0, stream>>>(
            xcur, agg, idg,
            wn0p + (size_t)i * 2 * Hdim * Hdim, nb0 + i * Hdim,
            wn1p + (size_t)i * Hdim * Hdim,     nb1 + i * Hdim,
            nlg + i * Hdim, nlb + i * Hdim, xnext);
        xcur = xnext;
    }
}
